// iEFLModule_46213848105560
// MI455X (gfx1250) — matrix operand order repaired and checked by reading, not compiled, not run on hardware
//
#include <hip/hip_runtime.h>
#include <hip/hip_bf16.h>
#include <math.h>

#define BS   32
#define EMBD 256
#define HW   4096
#define LSEQ 27
#define MLPD 2048

typedef __attribute__((ext_vector_type(16))) _Float16 v16h;
typedef __attribute__((ext_vector_type(8)))  _Float16 v8h;
typedef __attribute__((ext_vector_type(8)))  float    v8f;

__device__ __forceinline__ v8f wmma16(v16h a, v16h b, v8f c) {
  return __builtin_amdgcn_wmma_f32_16x16x32_f16(false, a, false, b, (short)0, c, false, false);
}

// A fragment: 16x32 f16, row-major source, lane r = row, hi selects K-half interleave.
__device__ __forceinline__ v16h load_a(const _Float16* row, int k0, int hi) {
  v8h lo = *(const v8h*)(row + k0 + hi * 8);
  v8h hh = *(const v8h*)(row + k0 + 16 + hi * 8);
  v16h r;
#pragma unroll
  for (int j = 0; j < 8; ++j) { r[j] = lo[j]; r[j + 8] = hh[j]; }
  return r;
}

// B fragment from N-major (transposed) storage: lane r = N; hi selects K-half interleave (same as A).
__device__ __forceinline__ v16h load_b(const _Float16* rowN, int k0, int hi) {
  v8h lo = *(const v8h*)(rowN + k0 + hi * 8);
  v8h hh = *(const v8h*)(rowN + k0 + 16 + hi * 8);
  v16h r;
#pragma unroll
  for (int j = 0; j < 8; ++j) { r[j] = lo[j]; r[j + 8] = hh[j]; }
  return r;
}

__device__ __forceinline__ float wave_max(float v) {
#pragma unroll
  for (int o = 16; o > 0; o >>= 1) v = fmaxf(v, __shfl_xor(v, o, 32));
  return v;
}
__device__ __forceinline__ float wave_sum(float v) {
#pragma unroll
  for (int o = 16; o > 0; o >>= 1) v += __shfl_xor(v, o, 32);
  return v;
}

// ---------------- query positional encoding table [27][256] ----------------
__global__ void k_qpe(float* qpe) {
  int c = threadIdx.x;  // 256
  for (int row = 0; row < 9; ++row) {
    int y = row / 3, x = row % 3;
    int k  = (c < 128) ? y : x;
    int cp = (c < 128) ? c : c - 128;
    int i = cp >> 1;
    float denom = powf(10000.0f, (float)i / 64.0f);
    float p = (float)k / denom;
    float val = (cp & 1) ? cosf(p) : sinf(p);
    for (int o = 0; o < 3; ++o) qpe[(o * 9 + row) * EMBD + c] = val;
  }
}

// ------------- f32 [K][F] -> f16 transposed [F][K] (N-major weights) -------------
__global__ void k_cvtT(const float* in, _Float16* out, int K, int F) {
  int id = blockIdx.x * blockDim.x + threadIdx.x;
  if (id < K * F) {
    int k = id / F, f = id % F;
    out[(size_t)f * K + k] = (_Float16)in[id];
  }
}

// ---------------- K/V projection: per-tile GEMM with WMMA ----------------
// grid (HW/64, EMBD/64, BS), block 128, dyn LDS = 3 * 64*264*2
__global__ void k_proj(const float* __restrict__ f_e, const float* __restrict__ pos,
                       const float* __restrict__ Wk, const float* __restrict__ bk,
                       const float* __restrict__ Wv, const float* __restrict__ bv,
                       _Float16* __restrict__ klog, _Float16* __restrict__ vout) {
  const int b = blockIdx.z;
  const int s0 = blockIdx.x * 64;
  const int f0 = blockIdx.y * 64;
  extern __shared__ char smem[];
  const int LD = 264;
  _Float16* kv = (_Float16*)smem;        // [64][264] tokens x emb
  _Float16* wk = kv + 64 * LD;           // [64][264] N-major Wk tile
  _Float16* wv = wk + 64 * LD;
  const int t = threadIdx.x;

  // pass 1: f_e (transposed source), coalesced along tokens
  for (int i = t; i < 64 * 256; i += 128) {
    int sl = i & 63, e = i >> 6;
    kv[sl * LD + e] = (_Float16)f_e[((size_t)b * EMBD + e) * HW + s0 + sl];
  }
  __syncthreads();
  // pass 2: add pos_emb, coalesced along emb
  for (int i = t; i < 64 * 256; i += 128) {
    int e = i & 255, sl = i >> 8;
    kv[sl * LD + e] = (_Float16)((float)kv[sl * LD + e] +
                                 pos[((size_t)b * HW + s0 + sl) * EMBD + e]);
  }
  // weight tiles, N-major
  for (int i = t; i < 64 * 256; i += 128) {
    int fl = i & 63, e = i >> 6;
    wk[fl * LD + e] = (_Float16)Wk[e * EMBD + f0 + fl];
    wv[fl * LD + e] = (_Float16)Wv[e * EMBD + f0 + fl];
  }
  __syncthreads();

  const int w = t >> 5, lane = t & 31, r = lane & 15, hi = lane >> 4;
  v8f kacc[4], vacc[4];
#pragma unroll
  for (int j = 0; j < 4; ++j) { kacc[j] = (v8f)0.0f; vacc[j] = (v8f)0.0f; }
  const _Float16* arow = kv + (16 * w + r) * LD;
  for (int k0 = 0; k0 < 256; k0 += 32) {
    v16h a = load_a(arow, k0, hi);
#pragma unroll
    for (int n = 0; n < 4; ++n) {
      v16h bK = load_b(wk + (16 * n + r) * LD, k0, hi);
      kacc[n] = wmma16(a, bK, kacc[n]);
      v16h bV = load_b(wv + (16 * n + r) * LD, k0, hi);
      vacc[n] = wmma16(a, bV, vacc[n]);
    }
  }
#pragma unroll
  for (int n = 0; n < 4; ++n) {
    int col = f0 + 16 * n + r;
    float bkv = bk[col], bvv = bv[col];
#pragma unroll
    for (int i = 0; i < 8; ++i) {
      int srow = s0 + 16 * w + i + 8 * hi;
      size_t off = ((size_t)b * HW + srow) * EMBD + col;
      klog[off] = (_Float16)(kacc[n][i] + bkv);
      vout[off] = (_Float16)(vacc[n][i] + bvv);
    }
  }
}

// ------- online column softmax statistics: max and 1/sum over 4096 tokens -------
__global__ void k_reduce(const _Float16* __restrict__ klog,
                         float* __restrict__ cmax, float* __restrict__ crinv) {
  int b = blockIdx.x, e = threadIdx.x;  // thread per column, coalesced along emb
  const _Float16* base = klog + (size_t)b * HW * EMBD + e;
  float m = -1e30f, Z = 0.0f;
  for (int s = 0; s < HW; ++s) {
    if ((s & 31) == 0)  // uniform branch: prefetch next token block
      __builtin_prefetch(base + (size_t)(s + 32) * EMBD, 0, 1);
    float x = (float)base[(size_t)s * EMBD];
    float mn = fmaxf(m, x);
    Z = Z * expf(m - mn) + expf(x - mn);
    m = mn;
  }
  cmax[b * EMBD + e] = m;
  crinv[b * EMBD + e] = 1.0f / Z;
}

// -------- context = softmax(k)^T @ v per batch; output ctxT[b][f][e] f16 --------
// grid (2, 2, BS) for 128x128 tiles, block 256; double-buffered 32-token chunks.
__global__ void k_context(const _Float16* __restrict__ klog, const _Float16* __restrict__ v,
                          const float* __restrict__ cmax, const float* __restrict__ crinv,
                          _Float16* __restrict__ ctxT) {
  const int b = blockIdx.z, e0 = blockIdx.x * 128, f0 = blockIdx.y * 128;
  __shared__ _Float16 At[2][128 * 40];  // [e][s] chunk, exp() applied
  __shared__ _Float16 Bt[2][128 * 40];  // [f][s] chunk
  const int t = threadIdx.x;            // 256
  const int w = t >> 5, lane = t & 31, r = lane & 15, hi = lane >> 4;
  const int eL = t & 127;               // staging column owned by this thread
  const int sb = t >> 7;                // 0/1 token phase
  const float me = cmax[b * EMBD + e0 + eL];
  const size_t baseA = (size_t)b * HW * EMBD + e0 + eL;
  const size_t baseB = (size_t)b * HW * EMBD + f0 + eL;

  v8f acc[8];
#pragma unroll
  for (int n = 0; n < 8; ++n) acc[n] = (v8f)0.0f;

  float av[16], bv[16];
  // stage chunk 0
#pragma unroll
  for (int j = 0; j < 16; ++j) {
    int s = sb + 2 * j;
    av[j] = (float)klog[baseA + (size_t)s * EMBD];
    bv[j] = (float)v[baseB + (size_t)s * EMBD];
  }
#pragma unroll
  for (int j = 0; j < 16; ++j) {
    int s = sb + 2 * j;
    At[0][eL * 40 + s] = (_Float16)expf(av[j] - me);
    Bt[0][eL * 40 + s] = (_Float16)bv[j];
  }

  const int NCH = HW / 32;
  for (int c = 0; c < NCH; ++c) {
    const int cur = c & 1, nxt = cur ^ 1;
    __syncthreads();
    if (c + 1 < NCH) {  // issue next chunk's global loads before the WMMA burst
      size_t so = (size_t)(c + 1) * 32 * EMBD;
#pragma unroll
      for (int j = 0; j < 16; ++j) {
        int s = sb + 2 * j;
        av[j] = (float)klog[baseA + so + (size_t)s * EMBD];
        bv[j] = (float)v[baseB + so + (size_t)s * EMBD];
      }
    }
    v16h af = load_a(&At[cur][(16 * w + r) * 40], 0, hi);
#pragma unroll
    for (int n = 0; n < 8; ++n) {
      v16h bb = load_b(&Bt[cur][(16 * n + r) * 40], 0, hi);
      acc[n] = wmma16(af, bb, acc[n]);
    }
    if (c + 1 < NCH) {
#pragma unroll
      for (int j = 0; j < 16; ++j) {
        int s = sb + 2 * j;
        At[nxt][eL * 40 + s] = (_Float16)expf(av[j] - me);
        Bt[nxt][eL * 40 + s] = (_Float16)bv[j];
      }
    }
  }

  float rv[8];
#pragma unroll
  for (int i = 0; i < 8; ++i) rv[i] = crinv[b * EMBD + e0 + 16 * w + i + 8 * hi];
#pragma unroll
  for (int n = 0; n < 8; ++n) {
    int f = f0 + 16 * n + r;
#pragma unroll
    for (int i = 0; i < 8; ++i) {
      int e = e0 + 16 * w + i + 8 * hi;
      ctxT[((size_t)b * EMBD + f) * EMBD + e] = (_Float16)(acc[n][i] * rv[i]);
    }
  }
}

// ---------------- row layernorm over 256, wave-per-row ----------------
__device__ __forceinline__ void ln_rows(float* F, const float* g, const float* beta,
                                        int w, int lane) {
  for (int rr = 0; rr < 4; ++rr) {
    int row = w + 8 * rr;
    float vals[8];
    float s = 0.0f;
    for (int c = lane, j = 0; c < 256; c += 32, ++j) { vals[j] = F[row * 256 + c]; s += vals[j]; }
    s = wave_sum(s);
    float m = s * (1.0f / 256.0f);
    float vs = 0.0f;
#pragma unroll
    for (int j = 0; j < 8; ++j) { float d = vals[j] - m; vs += d * d; }
    vs = wave_sum(vs);
    float inv = rsqrtf(vs * (1.0f / 256.0f) + 1e-5f);
    for (int c = lane, j = 0; c < 256; c += 32, ++j)
      F[row * 256 + c] = (vals[j] - m) * inv * g[c] + beta[c];
  }
}

// ---------------- decoder: one block per batch, 3 steps in LDS ----------------
// block 256, dyn LDS = 32*256*4 + 2*32*264*2
__global__ void k_decode(const float* __restrict__ proto, const float* __restrict__ qpe,
                         const _Float16* __restrict__ WqT, const float* __restrict__ bq,
                         const _Float16* __restrict__ ctxT,
                         const float* __restrict__ g2, const float* __restrict__ be2,
                         const _Float16* __restrict__ W1T, const float* __restrict__ b1,
                         const _Float16* __restrict__ W2T, const float* __restrict__ b2,
                         const float* __restrict__ g3, const float* __restrict__ be3,
                         float* __restrict__ out) {
  const int b = blockIdx.x;
  extern __shared__ char smem[];
  float*    F  = (float*)smem;                         // [32][256] f32
  _Float16* A  = (_Float16*)(smem + 32 * 256 * 4);     // [32][264] f16 operand buffer
  _Float16* Hb = A + 32 * 264;                         // [32][264] f16 scratch
  const int t = threadIdx.x;
  const int w = t >> 5, lane = t & 31, r = lane & 15, hi = lane >> 4;
  const int mt = w & 1, ntb = w >> 1;   // wave -> (row-tile, col-tile base)
  const _Float16* ctx = ctxT + (size_t)b * EMBD * EMBD;

  for (int i = t; i < 32 * 256; i += 256) {
    int row = i >> 8;
    F[i] = (row < LSEQ) ? proto[i] : 0.0f;
  }
  __syncthreads();

  for (int step = 0; step < 3; ++step) {
    // X = F + qpe -> A (f16)
    for (int i = t; i < 32 * 256; i += 256) {
      int row = i >> 8, e = i & 255;
      float x = F[i] + ((row < LSEQ) ? qpe[i] : 0.0f);
      A[row * 264 + e] = (_Float16)x;
    }
    __syncthreads();

    // q logits = (X @ Wq + bq) * scale  -> Hb
    {
      v8f acc[4];
#pragma unroll
      for (int j = 0; j < 4; ++j) acc[j] = (v8f)0.0f;
      const _Float16* arow = A + (16 * mt + r) * 264;
      for (int k0 = 0; k0 < 256; k0 += 32) {
        v16h af = load_a(arow, k0, hi);
#pragma unroll
        for (int j = 0; j < 4; ++j) {
          int nt = ntb + 4 * j;
          acc[j] = wmma16(af, load_b(WqT + (size_t)(16 * nt + r) * 256, k0, hi), acc[j]);
        }
      }
#pragma unroll
      for (int j = 0; j < 4; ++j) {
        int col = 16 * (ntb + 4 * j) + r;
        float bb = bq[col];
#pragma unroll
        for (int i = 0; i < 8; ++i) {
          int row = 16 * mt + i + 8 * hi;
          Hb[row * 264 + col] = (_Float16)((acc[j][i] + bb) * 0.0625f);
        }
      }
    }
    __syncthreads();

    // row softmax of logits -> q in A (f16)
    for (int rr = 0; rr < 4; ++rr) {
      int row = w + 8 * rr;
      float vals[8], vmax = -1e30f;
      for (int c = lane, j = 0; c < 256; c += 32, ++j) {
        vals[j] = (float)Hb[row * 264 + c];
        vmax = fmaxf(vmax, vals[j]);
      }
      vmax = wave_max(vmax);
      float Z = 0.0f;
#pragma unroll
      for (int j = 0; j < 8; ++j) { vals[j] = expf(vals[j] - vmax); Z += vals[j]; }
      Z = wave_sum(Z);
      float inv = 1.0f / Z;
      for (int c = lane, j = 0; c < 256; c += 32, ++j)
        A[row * 264 + c] = (_Float16)(vals[j] * inv);
    }
    __syncthreads();

    // att = q @ context ; F += att
    {
      v8f acc[4];
#pragma unroll
      for (int j = 0; j < 4; ++j) acc[j] = (v8f)0.0f;
      const _Float16* arow = A + (16 * mt + r) * 264;
      for (int k0 = 0; k0 < 256; k0 += 32) {
        v16h af = load_a(arow, k0, hi);
#pragma unroll
        for (int j = 0; j < 4; ++j) {
          int nt = ntb + 4 * j;
          acc[j] = wmma16(af, load_b(ctx + (size_t)(16 * nt + r) * 256, k0, hi), acc[j]);
        }
      }
#pragma unroll
      for (int j = 0; j < 4; ++j) {
        int col = 16 * (ntb + 4 * j) + r;
#pragma unroll
        for (int i = 0; i < 8; ++i) {
          int row = 16 * mt + i + 8 * hi;
          F[row * 256 + col] += acc[j][i];
        }
      }
    }
    __syncthreads();
    ln_rows(F, g2, be2, w, lane);
    __syncthreads();

    // A = f16(F)
    for (int i = t; i < 32 * 256; i += 256) A[(i >> 8) * 264 + (i & 255)] = (_Float16)F[i];
    __syncthreads();

    // ff = gelu(F@W1 + b1) @ W2, chunked over the 2048 hidden dim
    {
      v8f accf[4];
#pragma unroll
      for (int j = 0; j < 4; ++j) accf[j] = (v8f)0.0f;
      for (int ch = 0; ch < 8; ++ch) {
        v8f acc[4];
#pragma unroll
        for (int j = 0; j < 4; ++j) acc[j] = (v8f)0.0f;
        const _Float16* arow = A + (16 * mt + r) * 264;
        for (int k0 = 0; k0 < 256; k0 += 32) {
          v16h af = load_a(arow, k0, hi);
#pragma unroll
          for (int j = 0; j < 4; ++j) {
            int fg = ch * 256 + 16 * (ntb + 4 * j) + r;
            acc[j] = wmma16(af, load_b(W1T + (size_t)fg * 256, k0, hi), acc[j]);
          }
        }
        __syncthreads();  // prior chunk's Hb fully consumed
#pragma unroll
        for (int j = 0; j < 4; ++j) {
          int col = 16 * (ntb + 4 * j) + r;
          float bb = b1[ch * 256 + col];
#pragma unroll
          for (int i = 0; i < 8; ++i) {
            int row = 16 * mt + i + 8 * hi;
            float x = acc[j][i] + bb;
            Hb[row * 264 + col] = (_Float16)(0.5f * x * (1.0f + erff(x * 0.70710678118f)));
          }
        }
        __syncthreads();
        const _Float16* hrow = Hb + (16 * mt + r) * 264;
        for (int k0 = 0; k0 < 256; k0 += 32) {
          v16h af = load_a(hrow, k0, hi);
#pragma unroll
          for (int j = 0; j < 4; ++j) {
            int fo = 16 * (ntb + 4 * j) + r;
            accf[j] = wmma16(af, load_b(W2T + (size_t)fo * MLPD + ch * 256, k0, hi), accf[j]);
          }
        }
      }
#pragma unroll
      for (int j = 0; j < 4; ++j) {
        int col = 16 * (ntb + 4 * j) + r;
        float bb = b2[col];
#pragma unroll
        for (int i = 0; i < 8; ++i) {
          int row = 16 * mt + i + 8 * hi;
          F[row * 256 + col] += accf[j][i] + bb;
        }
      }
    }
    __syncthreads();
    ln_rows(F, g3, be3, w, lane);
    __syncthreads();

    // emit this step's F (first 27 rows)
    for (int i = t; i < LSEQ * 256; i += 256)
      out[((size_t)step * BS + b) * (LSEQ * 256) + i] = F[i];
    __syncthreads();
  }
}

extern "C" void kernel_launch(void* const* d_in, const int* in_sizes, int n_in,
                              void* d_out, int out_size, void* d_ws, size_t ws_size,
                              hipStream_t stream) {
  (void)in_sizes; (void)n_in; (void)out_size; (void)ws_size;
  const float* f_e  = (const float*)d_in[0];
  const float* pos  = (const float*)d_in[1];
  /* d_in[2] = bboxes (unused by reference) */
  const float* proto= (const float*)d_in[3];
  const float* Wq   = (const float*)d_in[4];
  const float* bq   = (const float*)d_in[5];
  const float* Wk   = (const float*)d_in[6];
  const float* bk   = (const float*)d_in[7];
  const float* Wv   = (const float*)d_in[8];
  const float* bv   = (const float*)d_in[9];
  const float* W1   = (const float*)d_in[10];
  const float* b1   = (const float*)d_in[11];
  const float* W2   = (const float*)d_in[12];
  const float* b2   = (const float*)d_in[13];
  const float* g2   = (const float*)d_in[14];
  const float* be2  = (const float*)d_in[15];
  const float* g3   = (const float*)d_in[16];
  const float* be3  = (const float*)d_in[17];
  float* out = (float*)d_out;

  char* ws = (char*)d_ws;
  size_t off = 0;
  auto carve = [&](size_t bytes) -> char* {
    char* p = ws + off;
    off = (off + bytes + 255) & ~(size_t)255;
    return p;
  };
  _Float16* klog = (_Float16*)carve((size_t)BS * HW * EMBD * 2);   // 64 MB (raw logits)
  _Float16* vbuf = (_Float16*)carve((size_t)BS * HW * EMBD * 2);   // 64 MB
  _Float16* ctxT = (_Float16*)carve((size_t)BS * EMBD * EMBD * 2); // 4 MB
  _Float16* WqT  = (_Float16*)carve((size_t)EMBD * EMBD * 2);
  _Float16* W1T  = (_Float16*)carve((size_t)EMBD * MLPD * 2);
  _Float16* W2T  = (_Float16*)carve((size_t)MLPD * EMBD * 2);
  float*    qpe  = (float*)carve((size_t)LSEQ * EMBD * 4);
  float*    cmax = (float*)carve((size_t)BS * EMBD * 4);
  float*    crinv= (float*)carve((size_t)BS * EMBD * 4);

  k_qpe<<<1, 256, 0, stream>>>(qpe);
  {
    int n = EMBD * EMBD;
    k_cvtT<<<(n + 255) / 256, 256, 0, stream>>>(Wq, WqT, EMBD, EMBD);
    n = EMBD * MLPD;
    k_cvtT<<<(n + 255) / 256, 256, 0, stream>>>(W1, W1T, EMBD, MLPD);
    n = MLPD * EMBD;
    k_cvtT<<<(n + 255) / 256, 256, 0, stream>>>(W2, W2T, MLPD, EMBD);
  }
  size_t smemProj = (size_t)3 * 64 * 264 * 2;  // 101376 B
  k_proj<<<dim3(HW / 64, EMBD / 64, BS), 128, smemProj, stream>>>(
      f_e, pos, Wk, bk, Wv, bv, klog, vbuf);
  k_reduce<<<BS, 256, 0, stream>>>(klog, cmax, crinv);
  k_context<<<dim3(2, 2, BS), 256, 0, stream>>>(klog, vbuf, cmax, crinv, ctxT);
  size_t smemDec = (size_t)32 * 256 * 4 + (size_t)2 * 32 * 264 * 2;  // 66560 B
  k_decode<<<BS, 256, smemDec, stream>>>(proto, qpe, WqT, bq, ctxT,
                                         g2, be2, W1T, b1, W2T, b2, g3, be3, out);
}
